// TreeOfThoughts_20375324852543
// MI455X (gfx1250) — matrix operand order repaired and checked by reading, not compiled, not run on hardware
//
#include <hip/hip_runtime.h>

// ---------------------------------------------------------------------------
// TreeOfThoughts expansion on MI455X (gfx1250, wave32, WMMA).
//   N=1024 thoughts, L=64, D=768, H=1024, V=32000, B=8 beams, K=4 kept.
// bf16 WMMA for all GEMMs. Weights converted+transposed to bf16 per call so
// every WMMA B-fragment is two contiguous 16B loads per lane; Wg2 in bf16
// (62.5MB) is L2-resident (192MB). GEMM: wave computes 16x64 (4 accs sharing
// one A fragment), A tile staged to LDS via global_load_async_to_lds_b128
// (ASYNCcnt) and read via ds_load_b128; K-loop is register double-buffered so
// WMMA math overlaps the in-flight B stream instead of waiting loadcnt==0.
// ---------------------------------------------------------------------------

#define NN    1024
#define LL    64
#define DD    768
#define HH    1024
#define VV    32000
#define BB    8
#define KTOP  4
#define INV_TEMP (1.0f / 0.7f)

typedef __attribute__((ext_vector_type(16))) __bf16        v16bf;
typedef __attribute__((ext_vector_type(8)))  float         v8f;
typedef __attribute__((ext_vector_type(4)))  unsigned int  u32x4;

union FragBF { v16bf bf; u32x4 u[2]; };

__device__ __forceinline__ unsigned short f32_to_bf16(float f) {
  unsigned int u = __float_as_uint(f);
  u += 0x7FFFu + ((u >> 16) & 1u);          // round-to-nearest-even
  return (unsigned short)(u >> 16);
}
__device__ __forceinline__ float bf16_to_f32(unsigned short h) {
  return __uint_as_float(((unsigned int)h) << 16);
}
__device__ __forceinline__ unsigned int mixu(unsigned int x) {
  x ^= x >> 16; x *= 0x7FEB352Du;
  x ^= x >> 15; x *= 0x846CA68Bu;
  x ^= x >> 16; return x;
}

// ---------------------------------------------------------------------------
// fp32 [Kd,Nc] -> bf16 [Nc,Kd]  (transpose so B fragments are contiguous in K)
// ---------------------------------------------------------------------------
__global__ void transpose_cvt_kernel(const float* __restrict__ src,
                                     unsigned short* __restrict__ dst,
                                     int Kd, int Nc) {
  size_t i = (size_t)blockIdx.x * blockDim.x + threadIdx.x;
  size_t total = (size_t)Kd * Nc;
  if (i >= total) return;
  int k = (int)(i / Nc);
  int n = (int)(i % Nc);
  dst[(size_t)n * Kd + k] = f32_to_bf16(src[i]);
}

// ---------------------------------------------------------------------------
// feat[n,d] = mean_l embeddings[n,l,d]  -> bf16
// ---------------------------------------------------------------------------
__global__ void mean_cvt_kernel(const float* __restrict__ emb,
                                unsigned short* __restrict__ feat) {
  int i = blockIdx.x * blockDim.x + threadIdx.x;
  if (i >= NN * DD) return;
  int n = i / DD, d = i % DD;
  const float* p = emb + (size_t)n * LL * DD + d;
  float s = 0.0f;
#pragma unroll 8
  for (int l = 0; l < LL; ++l) s += p[(size_t)l * DD];
  feat[i] = f32_to_bf16(s * (1.0f / (float)LL));
}

// ---------------------------------------------------------------------------
// bf16 WMMA GEMM: C[M,Nc] = epilogue(A[M,Kd] @ Wt[Nc,Kd]^T + bias)
//   epilogue: v = (acc + bias[col]) * scale;  if (relu) v = max(v,0)
// 256 threads = 8 waves, block tile 32(M) x 256(N); wave computes 16x64
// (4 N-tiles reusing one A fragment). A tile (32 x Kd, contiguous) staged to
// LDS via global_load_async_to_lds_b128; K-loop double-buffered in registers.
// Requires M%32==0, Nc%256==0, Kd%32==0. Dynamic LDS: 32*Kd*2 (<=64KB).
// ---------------------------------------------------------------------------
__global__ __launch_bounds__(256) void wmma_gemm_bf16(
    const unsigned short* __restrict__ A,
    const unsigned short* __restrict__ Wt,
    const float* __restrict__ bias,
    unsigned short* __restrict__ C,
    int M, int Kd, int Nc, float scale, int do_relu) {
  extern __shared__ unsigned short sA[];   // [32][Kd] row-major
  const int lane   = threadIdx.x & 31;
  const int wave   = threadIdx.x >> 5;     // 0..7
  const int mBlock = blockIdx.y * 32;
  const int nWave  = blockIdx.x * 256 + (wave & 3) * 64;  // wave's 64-col strip
  const int mWave  = (wave >> 2) * 16;                    // 0 or 16
  const int hsel   = lane >> 4;            // 0: lanes 0-15, 1: lanes 16-31
  const int l16    = lane & 15;

  // --- async-stage A tile: rows [mBlock, mBlock+32) are contiguous ---------
  {
    const char* gsrc = (const char*)(A + (size_t)mBlock * Kd);
    char*       ldst = (char*)sA;
    const int chunks = 4 * Kd;             // 16-byte chunks in 64*Kd bytes
    for (int c = threadIdx.x; c < chunks; c += 256) {
      unsigned           lds = (unsigned)(uintptr_t)(ldst + c * 16);
      unsigned long long ga  = (unsigned long long)(uintptr_t)(gsrc + c * 16);
      asm volatile("global_load_async_to_lds_b128 %0, %1, off"
                   :: "v"(lds), "v"(ga) : "memory");
    }
    asm volatile("s_wait_asynccnt 0" ::: "memory");
  }
  __syncthreads();

  // ISA 7.12.2 16-bit A layout: lane holds row M=lane%16; K chunks
  // [kd + 8*hsel .. +7] and [kd + 16 + 8*hsel .. +7]  -> read from LDS.
  const unsigned short* aPtr = sA + (size_t)(mWave + l16) * Kd + hsel * 8;
  // B (Wt[Nc,Kd] row-major): lane holds col N=lane%16; K chunks
  // [kd + 8*hsel .. +7] and [kd + 16 + 8*hsel .. +7] (same order as A).
  const unsigned short* bPtr[4];
#pragma unroll
  for (int j = 0; j < 4; ++j)
    bPtr[j] = Wt + (size_t)(nWave + j * 16 + l16) * Kd + hsel * 8;

  v8f acc[4];
  {
    v8f z = {};
#pragma unroll
    for (int j = 0; j < 4; ++j) acc[j] = z;
  }

  // --- prologue: load fragments for kd = 0 ---------------------------------
  FragBF aCur, bCur[4];
  aCur.u[0] = *(const u32x4*)(aPtr);
  aCur.u[1] = *(const u32x4*)(aPtr + 16);
#pragma unroll
  for (int j = 0; j < 4; ++j) {
    bCur[j].u[0] = *(const u32x4*)(bPtr[j]);
    bCur[j].u[1] = *(const u32x4*)(bPtr[j] + 16);
  }

  // --- main loop: issue loads for kd+32, then WMMA on kd -------------------
  for (int kd = 0; kd < Kd; kd += 32) {
    int kn = kd + 32;
    if (kn >= Kd) kn = 0;                  // wrap: dead prefetch, stays in-bounds
    FragBF aNxt, bNxt[4];
    aNxt.u[0] = *(const u32x4*)(aPtr + kn);
    aNxt.u[1] = *(const u32x4*)(aPtr + kn + 16);
#pragma unroll
    for (int j = 0; j < 4; ++j) {
      bNxt[j].u[0] = *(const u32x4*)(bPtr[j] + kn);
      bNxt[j].u[1] = *(const u32x4*)(bPtr[j] + kn + 16);
      __builtin_prefetch(bPtr[j] + kd + 256, 0, 1);   // global_prefetch_b8
    }
#pragma unroll
    for (int j = 0; j < 4; ++j)
      acc[j] = __builtin_amdgcn_wmma_f32_16x16x32_bf16(
          /*neg_a=*/false, aCur.bf, /*neg_b=*/false, bCur[j].bf,
          /*c_mod=*/(short)0, acc[j], /*reuse_a=*/false, /*reuse_b=*/false);
    aCur = aNxt;
#pragma unroll
    for (int j = 0; j < 4; ++j) bCur[j] = bNxt[j];
  }

  // C layout: VGPR r -> row (tile_m + r + 8*hsel), col (tile_n + lane%16).
#pragma unroll
  for (int j = 0; j < 4; ++j) {
    const int cCol = nWave + j * 16 + l16;
    const float bcol = bias[cCol];
#pragma unroll
    for (int r = 0; r < 8; ++r) {
      int row = mBlock + mWave + r + hsel * 8;
      float v = (acc[j][r] + bcol) * scale;
      if (do_relu) v = fmaxf(v, 0.0f);
      C[(size_t)row * Nc + cCol] = f32_to_bf16(v);
    }
  }
}

// ---------------------------------------------------------------------------
// Gumbel-max categorical sampling: tokens[n,b] = argmax_v logits[n,v]+g(n,b,v)
// logits already carry the 1/TEMP scale. One 256-thread block per thought.
// (Stateless hash RNG stands in for jax threefry; shapes/semantics preserved.)
// ---------------------------------------------------------------------------
__global__ __launch_bounds__(256) void sample_kernel(
    const unsigned short* __restrict__ logits, int* __restrict__ tokens) {
  const int n = blockIdx.x, tid = threadIdx.x;
  __shared__ float sv[256];
  __shared__ int   si[256];

  unsigned int keyb[BB];
#pragma unroll
  for (int b = 0; b < BB; ++b)
    keyb[b] = mixu((unsigned)n * 0x85EBCA6Bu + (unsigned)b * 0xC2B2AE35u + 0x9E3779B9u);

  float best[BB]; int bidx[BB];
#pragma unroll
  for (int b = 0; b < BB; ++b) { best[b] = -3.4e38f; bidx[b] = 0; }

  const unsigned short* row = logits + (size_t)n * VV;
  for (int v = tid; v < VV; v += 256) {
    float l = bf16_to_f32(row[v]);
#pragma unroll
    for (int b = 0; b < BB; ++b) {
      unsigned int h = mixu(keyb[b] ^ ((unsigned)v * 0x27D4EB2Fu));
      float u = (float)(h >> 8) * (1.0f / 16777216.0f) + (0.5f / 16777216.0f);
      float g = -__logf(-__logf(u));
      float val = l + g;
      if (val > best[b]) { best[b] = val; bidx[b] = v; }
    }
  }
  for (int b = 0; b < BB; ++b) {
    sv[tid] = best[b]; si[tid] = bidx[b];
    __syncthreads();
    for (int s = 128; s > 0; s >>= 1) {
      if (tid < s) {
        bool take = (sv[tid + s] > sv[tid]) ||
                    (sv[tid + s] == sv[tid] && si[tid + s] < si[tid]);
        if (take) { sv[tid] = sv[tid + s]; si[tid] = si[tid + s]; }
      }
      __syncthreads();
    }
    if (tid == 0) tokens[n * BB + b] = si[0];
    __syncthreads();
  }
}

// ---------------------------------------------------------------------------
// emb_bf16[nb, d] = bf16(embed_table[tokens[nb], d])
// ---------------------------------------------------------------------------
__global__ void gather_embed_kernel(const int* __restrict__ tokens,
                                    const float* __restrict__ table,
                                    unsigned short* __restrict__ emb) {
  int i = blockIdx.x * blockDim.x + threadIdx.x;
  if (i >= NN * BB * DD) return;
  int d = i % DD, nb = i / DD;
  int tok = tokens[nb];
  emb[i] = f32_to_bf16(table[(size_t)tok * DD + d]);
}

// ---------------------------------------------------------------------------
// scores[nb,0..2] = softmax(e2[nb,:] @ We3 + be3). One wave32 per row.
// ---------------------------------------------------------------------------
__global__ __launch_bounds__(256) void eval_head_kernel(
    const unsigned short* __restrict__ e2, const float* __restrict__ We3,
    const float* __restrict__ be3, float* __restrict__ scores, int rows) {
  int wave = (int)((blockIdx.x * (size_t)blockDim.x + threadIdx.x) >> 5);
  int lane = threadIdx.x & 31;
  if (wave >= rows) return;
  const unsigned short* rowp = e2 + (size_t)wave * 256;
  float a0 = 0.f, a1 = 0.f, a2 = 0.f;
  for (int i = lane; i < 256; i += 32) {
    float x = bf16_to_f32(rowp[i]);
    a0 += x * We3[i * 3 + 0];
    a1 += x * We3[i * 3 + 1];
    a2 += x * We3[i * 3 + 2];
  }
#pragma unroll
  for (int m = 16; m >= 1; m >>= 1) {
    a0 += __shfl_xor(a0, m, 32);
    a1 += __shfl_xor(a1, m, 32);
    a2 += __shfl_xor(a2, m, 32);
  }
  if (lane == 0) {
    a0 += be3[0]; a1 += be3[1]; a2 += be3[2];
    float mx = fmaxf(a0, fmaxf(a1, a2));
    float e0 = __expf(a0 - mx), e1 = __expf(a1 - mx), ee2 = __expf(a2 - mx);
    float inv = 1.0f / (e0 + e1 + ee2);
    scores[(size_t)wave * 3 + 0] = e0 * inv;
    scores[(size_t)wave * 3 + 1] = e1 * inv;
    scores[(size_t)wave * 3 + 2] = ee2 * inv;
  }
}

// ---------------------------------------------------------------------------
// Top-K=4 of B=8 by scores[...,0] (+const bonus -> order unchanged); emit
// selected raw score triples. One thread per thought.
// ---------------------------------------------------------------------------
__global__ void topk_kernel(const float* __restrict__ scores,
                            float* __restrict__ out) {
  int n = blockIdx.x * blockDim.x + threadIdx.x;
  if (n >= NN) return;
  const float* s = scores + (size_t)n * BB * 3;
  float s0[BB]; bool used[BB];
#pragma unroll
  for (int b = 0; b < BB; ++b) { s0[b] = s[b * 3]; used[b] = false; }
  for (int k = 0; k < KTOP; ++k) {
    int best = 0; float bv = -3.4e38f;
#pragma unroll
    for (int b = 0; b < BB; ++b)
      if (!used[b] && s0[b] > bv) { bv = s0[b]; best = b; }
    used[best] = true;
    size_t o = ((size_t)n * KTOP + k) * 3;
    out[o + 0] = s[best * 3 + 0];
    out[o + 1] = s[best * 3 + 1];
    out[o + 2] = s[best * 3 + 2];
  }
}

// ---------------------------------------------------------------------------
extern "C" void kernel_launch(void* const* d_in, const int* in_sizes, int n_in,
                              void* d_out, int out_size, void* d_ws, size_t ws_size,
                              hipStream_t stream) {
  const float* embeddings  = (const float*)d_in[0];
  const float* embed_table = (const float*)d_in[1];
  const float* Wg1 = (const float*)d_in[2];  const float* bg1 = (const float*)d_in[3];
  const float* Wg2 = (const float*)d_in[4];  const float* bg2 = (const float*)d_in[5];
  const float* We1 = (const float*)d_in[6];  const float* be1 = (const float*)d_in[7];
  const float* We2 = (const float*)d_in[8];  const float* be2 = (const float*)d_in[9];
  const float* We3 = (const float*)d_in[10]; const float* be3 = (const float*)d_in[11];

  char* ws = (char*)d_ws;
  size_t off = 0;
  auto alloc = [&](size_t bytes) -> void* {
    void* p = ws + off;
    off = (off + bytes + 255) & ~(size_t)255;
    return p;
  };
  unsigned short* wg1t   = (unsigned short*)alloc((size_t)HH * DD * 2);        // [H,D]
  unsigned short* wg2t   = (unsigned short*)alloc((size_t)VV * HH * 2);        // [V,H]
  unsigned short* we1t   = (unsigned short*)alloc((size_t)512 * DD * 2);       // [512,D]
  unsigned short* we2t   = (unsigned short*)alloc((size_t)256 * 512 * 2);      // [256,512]
  unsigned short* feat   = (unsigned short*)alloc((size_t)NN * DD * 2);        // [N,D]
  unsigned short* hbuf   = (unsigned short*)alloc((size_t)NN * HH * 2);        // [N,H]
  unsigned short* logits = (unsigned short*)alloc((size_t)NN * VV * 2);        // [N,V]
  int*            tokens = (int*)alloc((size_t)NN * BB * 4);                   // [N,B]
  unsigned short* embb   = (unsigned short*)alloc((size_t)NN * BB * DD * 2);   // [N*B,D]
  unsigned short* e1     = (unsigned short*)alloc((size_t)NN * BB * 512 * 2);  // [N*B,512]
  unsigned short* e2     = (unsigned short*)alloc((size_t)NN * BB * 256 * 2);  // [N*B,256]
  float*          scores = (float*)alloc((size_t)NN * BB * 3 * 4);             // [N*B,3]

  const int T = 256;
  auto cdiv = [](size_t a, size_t b) { return (unsigned)((a + b - 1) / b); };

  // 1) weight convert + transpose (fp32 -> bf16, [K,N] -> [N,K])
  transpose_cvt_kernel<<<cdiv((size_t)DD * HH, T), T, 0, stream>>>(Wg1, wg1t, DD, HH);
  transpose_cvt_kernel<<<cdiv((size_t)HH * VV, T), T, 0, stream>>>(Wg2, wg2t, HH, VV);
  transpose_cvt_kernel<<<cdiv((size_t)DD * 512, T), T, 0, stream>>>(We1, we1t, DD, 512);
  transpose_cvt_kernel<<<cdiv((size_t)512 * 256, T), T, 0, stream>>>(We2, we2t, 512, 256);

  // 2) mean over tokens -> bf16 features
  mean_cvt_kernel<<<cdiv((size_t)NN * DD, T), T, 0, stream>>>(embeddings, feat);

  // 3) h = relu(feat @ Wg1 + bg1)            [1024 x 1024, K=768]
  wmma_gemm_bf16<<<dim3(HH / 256, NN / 32), T, 32 * DD * 2, stream>>>(
      feat, wg1t, bg1, hbuf, NN, DD, HH, 1.0f, 1);

  // 4) logits = (h @ Wg2 + bg2) / TEMP       [1024 x 32000, K=1024]  (dominant)
  wmma_gemm_bf16<<<dim3(VV / 256, NN / 32), T, 32 * HH * 2, stream>>>(
      hbuf, wg2t, bg2, logits, NN, HH, VV, INV_TEMP, 0);

  // 5) Gumbel-max categorical sampling -> tokens [N,B]
  sample_kernel<<<NN, T, 0, stream>>>(logits, tokens);

  // 6) gather token embeddings -> bf16 [N*B, D]
  gather_embed_kernel<<<cdiv((size_t)NN * BB * DD, T), T, 0, stream>>>(
      tokens, embed_table, embb);

  // 7) evaluator MLP: relu(.@We1+be1) -> relu(.@We2+be2)
  wmma_gemm_bf16<<<dim3(512 / 256, (NN * BB) / 32), T, 32 * DD * 2, stream>>>(
      embb, we1t, be1, e1, NN * BB, DD, 512, 1.0f, 1);
  wmma_gemm_bf16<<<dim3(256 / 256, (NN * BB) / 32), T, 32 * 512 * 2, stream>>>(
      e1, we2t, be2, e2, NN * BB, 512, 256, 1.0f, 1);

  // 8) final 3-way head + softmax (one wave per row)
  eval_head_kernel<<<(NN * BB) / 8, T, 0, stream>>>(e2, We3, be3, scores, NN * BB);

  // 9) top-4 of 8 beams, emit selected score triples -> d_out [N,4,3] fp32
  topk_kernel<<<cdiv(NN, T), T, 0, stream>>>(scores, (float*)d_out);
}
